// ContrastiveGNN_84963043049901
// MI455X (gfx1250) — matrix operand order repaired and checked by reading, not compiled, not run on hardware
//
#include <hip/hip_runtime.h>
#include <hip/hip_bf16.h>

#define NN 100000
#define EE 600000
#define DD 128
#define CC 40
#define EPSV 1e-5f
#define WSTRIDE 136    // padded LDS row stride in halves (272 B): bank-conflict-free b128 loads
#define PADROWS 128    // bufA padding rows so GEMM stores need no bounds checks

typedef __attribute__((ext_vector_type(16))) _Float16 v16h;
typedef __attribute__((ext_vector_type(8)))  _Float16 v8h;
typedef __attribute__((ext_vector_type(8)))  float    v8f;

// ---------------- degree / normalization ----------------
__global__ void deg_init_k(int* __restrict__ deg) {
    int i = blockIdx.x * blockDim.x + threadIdx.x;
    if (i < NN) deg[i] = 1;   // self-loop
}

__global__ void deg_count_k(const int* __restrict__ ei, int* __restrict__ deg) {
    int e = blockIdx.x * blockDim.x + threadIdx.x;
    if (e < EE) atomicAdd(&deg[ei[EE + e]], 1);   // dst side
}

__global__ void dinv_k(float* __restrict__ dinv) {
    int i = blockIdx.x * blockDim.x + threadIdx.x;
    if (i < NN) {
        int d = ((const int*)dinv)[i];
        dinv[i] = rsqrtf((float)d);
    }
}

// ---------------- WMMA GEMM: O[NN x 128] = H[NN x 128] @ W[128 x 128] (+bias, +relu) ----------------
// O must have >= gridDim*128 rows of backing store (bufA is padded); H rows are clamped.
template <bool HASBIAS, bool RELU>
__launch_bounds__(256)
__global__ void gemm128_k(const float* __restrict__ H, const float* __restrict__ W,
                          const float* __restrict__ bias, float* __restrict__ O) {
    // W transposed in LDS: Wt[n][k], row stride 136 halves (272 B)
    __shared__ _Float16 Wt[DD * WSTRIDE];   // ~34 KB
    int tid = threadIdx.x;
    for (int idx = tid; idx < DD * DD; idx += 256) {
        int k = idx >> 7, n = idx & 127;    // consecutive tid -> consecutive n: coalesced read
        Wt[n * WSTRIDE + k] = (_Float16)W[idx];
    }
    __syncthreads();

    int wave = tid >> 5, lane = tid & 31;
    int hg = lane >> 4, lm = lane & 15;     // half-wave group, lane-in-16
    int rowbase = blockIdx.x * 128 + wave * 16;
    int m = rowbase + lm;
    int mclamp = m < NN ? m : (NN - 1);     // OOB rows read in-bounds garbage; never stored/consumed
    const float* hrow = H + (size_t)mclamp * DD;

    // A fragments: 16x32 f16 per K-chunk (ISA 7.12.2 layout), unconditional float4 loads
    v16h a[4];
#pragma unroll
    for (int kc = 0; kc < 4; ++kc) {
        int kb = kc * 32 + hg * 8;
        float4 q0 = *(const float4*)(hrow + kb);
        float4 q1 = *(const float4*)(hrow + kb + 4);
        float4 q2 = *(const float4*)(hrow + kb + 16);
        float4 q3 = *(const float4*)(hrow + kb + 20);
        a[kc][0]  = (_Float16)q0.x; a[kc][1]  = (_Float16)q0.y;
        a[kc][2]  = (_Float16)q0.z; a[kc][3]  = (_Float16)q0.w;
        a[kc][4]  = (_Float16)q1.x; a[kc][5]  = (_Float16)q1.y;
        a[kc][6]  = (_Float16)q1.z; a[kc][7]  = (_Float16)q1.w;
        a[kc][8]  = (_Float16)q2.x; a[kc][9]  = (_Float16)q2.y;
        a[kc][10] = (_Float16)q2.z; a[kc][11] = (_Float16)q2.w;
        a[kc][12] = (_Float16)q3.x; a[kc][13] = (_Float16)q3.y;
        a[kc][14] = (_Float16)q3.z; a[kc][15] = (_Float16)q3.w;
    }

#pragma unroll
    for (int nt = 0; nt < 8; ++nt) {
        int n = nt * 16 + lm;
        const _Float16* wrow = Wt + (size_t)n * WSTRIDE;
        float cb = HASBIAS ? bias[n] : 0.0f;
        v8f c = {cb, cb, cb, cb, cb, cb, cb, cb};
#pragma unroll
        for (int kc = 0; kc < 4; ++kc) {
            // B fragment: halves k=8h..8h+7 and 16+8h..16+8h+7 -> two ds_load_b128
            union { v16h v; v8h h[2]; } bu;
            const _Float16* bp = wrow + kc * 32 + hg * 8;
            bu.h[0] = *(const v8h*)(bp);
            bu.h[1] = *(const v8h*)(bp + 16);
            c = __builtin_amdgcn_wmma_f32_16x16x32_f16(false, a[kc], false, bu.v,
                                                       (short)0, c, false, false);
        }
        // epilogue: one base pointer, constant-offset stores, no guards (O is padded)
        float* op = O + (size_t)(rowbase + hg * 8) * DD + n;
#pragma unroll
        for (int r = 0; r < 8; ++r) {
            float v = c[r];
            if (RELU) v = fmaxf(v, 0.0f);
            op[r * DD] = v;
        }
    }
}

// ---------------- aggregation ----------------
// h[i] = bias + dinv[i]^2 * xw[i]   (self-loop term)
__global__ void agg_init_k(const float* __restrict__ xw, const float* __restrict__ dinv,
                           const float* __restrict__ bias, float* __restrict__ h) {
    int idx = blockIdx.x * blockDim.x + threadIdx.x;   // over N*32 float4s
    if (idx >= NN * 32) return;
    int i = idx >> 5, q = (idx & 31) * 4;
    float di = dinv[i];
    float w = di * di;
    float4 x = *(const float4*)(xw + (size_t)i * DD + q);
    float4 r;
    r.x = bias[q + 0] + w * x.x;
    r.y = bias[q + 1] + w * x.y;
    r.z = bias[q + 2] + w * x.z;
    r.w = bias[q + 3] + w * x.w;
    *(float4*)(h + (size_t)i * DD + q) = r;
}

// one wave per edge: edge id is wave-uniform -> scalar loads of src/dst/dinv
__launch_bounds__(256)
__global__ void agg_edge_k(const int* __restrict__ ei, const float* __restrict__ xw,
                           const float* __restrict__ dinv, float* __restrict__ h) {
    int lane = threadIdx.x & 31;
    int e = __builtin_amdgcn_readfirstlane(blockIdx.x * 8 + (threadIdx.x >> 5));
    if (e >= EE) return;
    int s = ei[e], d = ei[EE + e];
    float w = dinv[s] * dinv[d];
    float4 x = *(const float4*)(xw + (size_t)s * DD + lane * 4);
    float* hp = h + (size_t)d * DD + lane * 4;
    unsafeAtomicAdd(hp + 0, w * x.x);
    unsafeAtomicAdd(hp + 1, w * x.y);
    unsafeAtomicAdd(hp + 2, w * x.z);
    unsafeAtomicAdd(hp + 3, w * x.w);
}

// ---------------- batch norm ----------------
__global__ void zero_stats_k(float* __restrict__ g) {
    g[threadIdx.x] = 0.0f;   // 256 floats: gsum[128] ++ gsq[128]
}

__launch_bounds__(256)
__global__ void bn_stats_k(const float* __restrict__ h, float* __restrict__ gsum,
                           float* __restrict__ gsq) {
    __shared__ float s0[256], s1[256];
    int t = threadIdx.x;
    int c = t & 127, half = t >> 7;
    int base = blockIdx.x * 128;
    float sum = 0.0f, sq = 0.0f;
    int rend = base + 128;
    if (rend > NN) rend = NN;
    for (int r = base + half; r < rend; r += 2) {
        float v = h[(size_t)r * DD + c];
        sum += v;
        sq += v * v;
    }
    s0[t] = sum; s1[t] = sq;
    __syncthreads();
    if (half == 0) {
        unsafeAtomicAdd(&gsum[c], s0[t] + s0[t + 128]);
        unsafeAtomicAdd(&gsq[c],  s1[t] + s1[t + 128]);
    }
}

__global__ void bn_apply_k(float* __restrict__ h, const float* __restrict__ gsum,
                           const float* __restrict__ gsq, const float* __restrict__ gamma,
                           const float* __restrict__ beta, const float* __restrict__ pa) {
    int idx = blockIdx.x * blockDim.x + threadIdx.x;   // over N*32 float4s
    if (idx >= NN * 32) return;
    int i = idx >> 5, q = (idx & 31) * 4;
    float aslope = pa[0];
    float4 v = *(float4*)(h + (size_t)i * DD + q);
    float vv[4] = {v.x, v.y, v.z, v.w};
    float oo[4];
#pragma unroll
    for (int j = 0; j < 4; ++j) {
        int cj = q + j;
        float mu = gsum[cj] * (1.0f / NN);
        float var = gsq[cj] * (1.0f / NN) - mu * mu;
        float rstd = rsqrtf(var + EPSV);
        float y = (vv[j] - mu) * rstd * gamma[cj] + beta[cj];
        oo[j] = y > 0.0f ? y : aslope * y;
    }
    float4 r = {oo[0], oo[1], oo[2], oo[3]};
    *(float4*)(h + (size_t)i * DD + q) = r;
}

// ---------------- head: logits = hid @ Wh2 + bh2 ; argmax ----------------
__launch_bounds__(256)
__global__ void head2_k(const float* __restrict__ hid, const float* __restrict__ W2,
                        const float* __restrict__ b2, float* __restrict__ logits,
                        float* __restrict__ outi) {
    __shared__ float w[DD * CC];   // 20 KB, [k][c]
    int t = threadIdx.x;
    for (int idx = t; idx < DD * CC; idx += 256) w[idx] = W2[idx];
    __syncthreads();
    int i = blockIdx.x * 256 + t;
    if (i >= NN) return;
    float acc[CC];
#pragma unroll
    for (int c = 0; c < CC; ++c) acc[c] = b2[c];
    const float* row = hid + (size_t)i * DD;
    for (int k = 0; k < DD; ++k) {
        float hv = row[k];
#pragma unroll
        for (int c = 0; c < CC; ++c) acc[c] = fmaf(hv, w[k * CC + c], acc[c]);
    }
    float best = acc[0];
    int bi = 0;
#pragma unroll
    for (int c = 1; c < CC; ++c)
        if (acc[c] > best) { best = acc[c]; bi = c; }
    float* lg = logits + (size_t)i * CC;
#pragma unroll
    for (int c = 0; c < CC; ++c) lg[c] = acc[c];
    outi[i] = (float)bi;
}

// ---------------- driver ----------------
extern "C" void kernel_launch(void* const* d_in, const int* in_sizes, int n_in,
                              void* d_out, int out_size, void* d_ws, size_t ws_size,
                              hipStream_t stream) {
    const float* x      = (const float*)d_in[0];
    const int*   ei     = (const int*)d_in[1];
    const float* Ws     = (const float*)d_in[2];
    const float* bs     = (const float*)d_in[3];
    const float* gammas = (const float*)d_in[4];
    const float* betas  = (const float*)d_in[5];
    const float* pa     = (const float*)d_in[6];
    const float* Wh1    = (const float*)d_in[7];
    const float* bh1    = (const float*)d_in[8];
    const float* Wh2    = (const float*)d_in[9];
    const float* bh2    = (const float*)d_in[10];

    float* emb    = (float*)d_out;                 // N*D, doubles as persistent h buffer
    float* logits = emb + (size_t)NN * DD;         // N*C
    float* outi   = logits + (size_t)NN * CC;      // N

    // ws layout: small arrays first, then bufA (padded by PADROWS rows) last
    float* dinv = (float*)d_ws;                    // N (int degree, then float dinv)
    float* gsum = dinv + NN;                       // 128
    float* gsq  = gsum + DD;                       // 128
    float* bufA = gsq + DD;                        // (N + PADROWS) * D scratch (xw / hidden)

    deg_init_k<<<(NN + 255) / 256, 256, 0, stream>>>((int*)dinv);
    deg_count_k<<<(EE + 255) / 256, 256, 0, stream>>>(ei, (int*)dinv);
    dinv_k<<<(NN + 255) / 256, 256, 0, stream>>>(dinv);

    int gemm_blocks = (NN + 127) / 128;            // 782; writes up to 100096 rows (< NN+PADROWS)
    int el4_blocks  = (NN * 32 + 255) / 256;
    int edge_blocks = (EE + 7) / 8;                // one wave (32 lanes) per edge

    for (int l = 0; l < 3; ++l) {
        const float* hin = (l == 0) ? x : (const float*)emb;
        gemm128_k<false, false><<<gemm_blocks, 256, 0, stream>>>(
            hin, Ws + (size_t)l * DD * DD, nullptr, bufA);
        agg_init_k<<<el4_blocks, 256, 0, stream>>>(bufA, dinv, bs + (size_t)l * DD, emb);
        agg_edge_k<<<edge_blocks, 256, 0, stream>>>(ei, bufA, dinv, emb);
        zero_stats_k<<<1, 256, 0, stream>>>(gsum);
        bn_stats_k<<<gemm_blocks, 256, 0, stream>>>(emb, gsum, gsq);
        bn_apply_k<<<el4_blocks, 256, 0, stream>>>(emb, gsum, gsq,
                                                   gammas + (size_t)l * DD,
                                                   betas + (size_t)l * DD, pa + l);
    }

    // head: relu(emb @ Wh1 + bh1) -> bufA ; bufA @ Wh2 + bh2 -> logits ; argmax -> outi
    gemm128_k<true, true><<<gemm_blocks, 256, 0, stream>>>(emb, Wh1, bh1, bufA);
    head2_k<<<(NN + 255) / 256, 256, 0, stream>>>(bufA, Wh2, bh2, logits, outi);
}
